// TrapezoidAttention_38525856645408
// MI455X (gfx1250) — matrix operand order repaired and checked by reading, not compiled, not run on hardware
//
#include <hip/hip_runtime.h>
#include <hip/hip_bf16.h>

typedef __attribute__((ext_vector_type(16))) _Float16 v16h;
typedef __attribute__((ext_vector_type(8)))  _Float16 v8h;
typedef __attribute__((ext_vector_type(8)))  float    v8f;

#define BB 16
#define CC 64
#define HH 256
#define WW 256

// ---- workspace layout (bytes) ----
#define WS_ATT   0u
#define WS_POOL  (16u * 256u * 256u * 4u)       // att: B*H*W f32
#define WS_DIR   (WS_POOL + 1024u * 4u)         // pooled: B*C f32
#define WS_WPACK (WS_DIR + 64u)                 // wpack: 73,728 B (32B aligned)

// ------------------------------------------------------------------
// Pack conv weights (f32 OIHW) into per-lane WMMA B-fragment layout:
// fragment f = tap(dh,dw)*2 + kchunk ; per lane: N = lane%16 (out ch),
// K = j + 16*(lane>=16) (in-ch within 32-chunk). 16 contiguous halves
// per lane per fragment -> one 32B load in the GEMM kernel.
// ------------------------------------------------------------------
__global__ void pack_w_kernel(const float* __restrict__ conv_w,
                              _Float16* __restrict__ wp) {
    int idx = blockIdx.x * 256 + threadIdx.x;      // 0 .. 36863
    int j    = idx & 15;
    int lane = (idx >> 4) & 31;
    int fn   = idx >> 9;                           // n*18 + f
    int f    = fn % 18;
    int n    = fn / 18;
    int t    = f >> 1;                             // tap 0..8
    int kc   = f & 1;                              // K chunk 0/1
    int dh   = t / 3, dw = t % 3;
    int oc   = n * 16 + (lane & 15);
    int ic   = kc * 32 + j + ((lane & 16) ? 16 : 0);
    wp[idx] = (_Float16)conv_w[(oc * CC + ic) * 9 + dh * 3 + dw];
}

// ------------------------------------------------------------------
// Global mean pool of x over (H,W): one block per (b,c).
// ------------------------------------------------------------------
__global__ void pool_kernel(const float* __restrict__ x,
                            float* __restrict__ pooled) {
    int bc = blockIdx.x;                           // 0..1023
    const float4* xp = (const float4*)(x + (size_t)bc * (HH * WW));
    float s = 0.f;
    for (int i = threadIdx.x; i < HH * WW / 4; i += 256) {
        float4 v = xp[i];
        s += v.x + v.y + v.z + v.w;
    }
#pragma unroll
    for (int off = 16; off; off >>= 1) s += __shfl_xor(s, off, 32);
    __shared__ float ls[8];
    int lane = threadIdx.x & 31, wv = threadIdx.x >> 5;
    if (lane == 0) ls[wv] = s;
    __syncthreads();
    if (threadIdx.x == 0) {
        float t = 0.f;
#pragma unroll
        for (int i = 0; i < 8; ++i) t += ls[i];
        pooled[bc] = t * (1.f / (float)(HH * WW));
    }
}

// ------------------------------------------------------------------
// direction MLP: pooled[b,:] -> relu(64->8) -> sigmoid(8->1).
// ------------------------------------------------------------------
__global__ void dir_kernel(const float* __restrict__ pooled,
                           const float* __restrict__ d1w,
                           const float* __restrict__ d1b,
                           const float* __restrict__ d2w,
                           const float* __restrict__ d2b,
                           float* __restrict__ dir_ws,
                           float* __restrict__ dir_out) {
    int b = threadIdx.x;
    if (b >= BB) return;
    float z = d2b[0];
#pragma unroll
    for (int j = 0; j < 8; ++j) {
        float h1 = d1b[j];
        for (int c = 0; c < CC; ++c) h1 += pooled[b * CC + c] * d1w[j * CC + c];
        h1 = h1 > 0.f ? h1 : 0.f;
        z += h1 * d2w[j];
    }
    float d = 1.f / (1.f + expf(-z));
    dir_ws[b]  = d;
    dir_out[b] = d;   // direction output (B,1,1,1)
}

// ------------------------------------------------------------------
// Fused implicit-GEMM conv3x3 (WMMA f16->f32) + bias + BN + ReLU +
// attn 1x1 reduction. Block tile: 16 pixels (W) x 4 rows (H) x 64 oc.
// 256 threads = 8 waves: wave = (row-pair rp 0..1) x (oc-tile 0..3).
// Each wave: 2 accumulators x 18 fragments = 36 WMMAs, B-fragment
// held in registers and reused across both row accumulators.
// ------------------------------------------------------------------
__global__ void __launch_bounds__(256) conv_att_kernel(
    const float* __restrict__ x, const _Float16* __restrict__ wp,
    const float* __restrict__ conv_b, const float* __restrict__ gamma,
    const float* __restrict__ beta, const float* __restrict__ mean,
    const float* __restrict__ var, const float* __restrict__ attn_w,
    const float* __restrict__ attn_b, float* __restrict__ att) {

    __shared__ alignas(32) _Float16 xs[6 * 18 * 64]; // [row][w_local][c] f16
    __shared__ float attp[4][4][16];                 // [oc_tile][row][pixel]

    const int tid = threadIdx.x;
    const int w0 = blockIdx.x * 16;
    const int h0 = blockIdx.y * 4;
    const int b  = blockIdx.z;

    // Stage 6 input rows (4 output rows + halo) to LDS, transposed so
    // channels are contiguous per pixel (channels map directly onto K).
    for (int e = tid; e < 6 * 64 * 18; e += 256) {
        int r   = e / (64 * 18);
        int rem = e - r * (64 * 18);
        int c   = rem / 18;
        int wl  = rem - c * 18;
        int gh = h0 + r - 1, gw = w0 + wl - 1;
        float v = 0.f;
        if ((unsigned)gh < HH && (unsigned)gw < WW)
            v = x[(((b * CC + c) * HH + gh) << 8) + gw];
        xs[(r * 18 + wl) * 64 + c] = (_Float16)v;
    }
    __syncthreads();

    const int lane = tid & 31;
    const int ww   = tid >> 5;
    const int oct  = ww & 3;            // out-channel tile
    const int rp   = ww >> 2;           // row pair: rows {2rp, 2rp+1}
    const int m    = lane & 15;         // A-matrix row (pixel within 16)
    const int chi  = (lane & 16) ? 16 : 0;

    v8f acc0 = {};
    v8f acc1 = {};
    const _Float16* wbase = wp + (size_t)(oct * 18) * 512 + lane * 16;

#pragma unroll
    for (int f = 0; f < 18; ++f) {
        const int tap = f >> 1, kc = f & 1;
        const int tdh = tap / 3, tdw = tap % 3;
        // B fragment: prepacked, one 32B contiguous load per lane (L2 resident).
        v16h bm = *(const v16h*)(wbase + f * 512);
        // A fragments for the two output rows of this wave.
        const _Float16* ap0 =
            &xs[((rp * 2 + tdh) * 18 + m + tdw) * 64 + kc * 32 + chi];
        const _Float16* ap1 = ap0 + 18 * 64;
        v16h a0, a1;
        ((v8h*)&a0)[0] = *(const v8h*)ap0;
        ((v8h*)&a0)[1] = *(const v8h*)(ap0 + 8);
        ((v8h*)&a1)[0] = *(const v8h*)ap1;
        ((v8h*)&a1)[1] = *(const v8h*)(ap1 + 8);
        acc0 = __builtin_amdgcn_wmma_f32_16x16x32_f16(
            false, a0, false, bm, (short)0, acc0, false, false);
        acc1 = __builtin_amdgcn_wmma_f32_16x16x32_f16(
            false, a1, false, bm, (short)0, acc1, false, false);
    }

    // Epilogue: bias + BN + ReLU, then dot with attn_w across N (out ch).
    const int oc = oct * 16 + m;                    // N = lane%16
    const float sfac = gamma[oc] * rsqrtf(var[oc] + 1e-5f);
    const float tfac = (conv_b[oc] - mean[oc]) * sfac + beta[oc];
    const float aw = attn_w[oc];
    const int mh = (lane & 16) ? 8 : 0;

#pragma unroll
    for (int half = 0; half < 2; ++half) {
        float p[8];
#pragma unroll
        for (int r = 0; r < 8; ++r) {
            float v = (half ? acc1[r] : acc0[r]) * sfac + tfac;
            v = v > 0.f ? v : 0.f;
            p[r] = v * aw;
        }
#pragma unroll
        for (int r = 0; r < 8; ++r) {
#pragma unroll
            for (int off = 8; off; off >>= 1)
                p[r] += __shfl_xor(p[r], off, 32);  // reduce over 16-lane N group
        }
        if ((lane & 15) == 0) {                     // lanes 0 (M=0..7), 16 (M=8..15)
#pragma unroll
            for (int r = 0; r < 8; ++r)
                attp[oct][rp * 2 + half][mh + r] = p[r];
        }
    }
    __syncthreads();
    if (tid < 64) {
        int row = tid >> 4, pix = tid & 15;
        float s = attn_b[0] + attp[0][row][pix] + attp[1][row][pix] +
                  attp[2][row][pix] + attp[3][row][pix];
        att[((b * HH + h0 + row) << 8) + w0 + pix] = s;  // pre-mask, pre-sigmoid
    }
}

// ------------------------------------------------------------------
// out = x * sigmoid(att * mask(b,h)); float4 streaming.
// ------------------------------------------------------------------
__global__ void final_kernel(const float* __restrict__ x,
                             const float* __restrict__ att,
                             const float* __restrict__ dir,
                             float* __restrict__ out) {
    size_t q = (size_t)blockIdx.x * 256 + threadIdx.x;  // float4 index
    size_t e = q * 4;
    int w = (int)(e & 255);
    int h = (int)((e >> 8) & 255);
    int b = (int)(e >> 22);                              // 64*256*256 = 2^22 per batch
    float jn = (float)h * (1.f / (float)HH);
    float mk = (dir[b] > 0.5f) ? (0.5f + 0.5f * jn) : (1.f - 0.5f * jn);
    float4 xv = *(const float4*)(x + e);
    float4 av = *(const float4*)(att + (((size_t)(b * HH + h)) << 8) + w);
    float4 o;
    o.x = xv.x / (1.f + __expf(-av.x * mk));
    o.y = xv.y / (1.f + __expf(-av.y * mk));
    o.z = xv.z / (1.f + __expf(-av.z * mk));
    o.w = xv.w / (1.f + __expf(-av.w * mk));
    *(float4*)(out + e) = o;
}

extern "C" void kernel_launch(void* const* d_in, const int* in_sizes, int n_in,
                              void* d_out, int out_size, void* d_ws, size_t ws_size,
                              hipStream_t stream) {
    const float* x      = (const float*)d_in[0];
    const float* conv_w = (const float*)d_in[1];
    const float* conv_b = (const float*)d_in[2];
    const float* gamma  = (const float*)d_in[3];
    const float* beta   = (const float*)d_in[4];
    const float* mean   = (const float*)d_in[5];
    const float* var    = (const float*)d_in[6];
    const float* attn_w = (const float*)d_in[7];
    const float* attn_b = (const float*)d_in[8];
    const float* d1w    = (const float*)d_in[9];
    const float* d1b    = (const float*)d_in[10];
    const float* d2w    = (const float*)d_in[11];
    const float* d2b    = (const float*)d_in[12];

    float* out = (float*)d_out;
    char*  ws  = (char*)d_ws;
    float*     att    = (float*)(ws + WS_ATT);
    float*     pooled = (float*)(ws + WS_POOL);
    float*     dirb   = (float*)(ws + WS_DIR);
    _Float16*  wp     = (_Float16*)(ws + WS_WPACK);

    // weights -> WMMA fragment layout
    pack_w_kernel<<<144, 256, 0, stream>>>(conv_w, wp);
    // pooled mean + direction MLP
    pool_kernel<<<BB * CC, 256, 0, stream>>>(x, pooled);
    dir_kernel<<<1, 32, 0, stream>>>(pooled, d1w, d1b, d2w, d2b, dirb,
                                     out + (size_t)BB * CC * HH * WW);
    // fused WMMA conv + BN + ReLU + attn reduction -> att[b,h,w]
    conv_att_kernel<<<dim3(WW / 16, HH / 4, BB), 256, 0, stream>>>(
        x, wp, conv_b, gamma, beta, mean, var, attn_w, attn_b, att);
    // elementwise gate
    final_kernel<<<(size_t)BB * CC * HH * WW / 4 / 256, 256, 0, stream>>>(
        x, att, dirb, out);
}
